// DockingAwareAttention_71262097376079
// MI455X (gfx1250) — matrix operand order repaired and checked by reading, not compiled, not run on hardware
//
#include <hip/hip_runtime.h>

// ---------------------------------------------------------------------------
// DockingAwareAttention for MI455X (gfx1250, wave32, WMMA bf16)
// B=4, S=2048, D=512, H=8, HD=64
// Register-blocked: GEMMs 16x64 per wave, attention 32 queries per wave.
// ---------------------------------------------------------------------------

typedef __attribute__((ext_vector_type(16))) __bf16 v16bf;
typedef __attribute__((ext_vector_type(8)))  __bf16 v8bf;
typedef __attribute__((ext_vector_type(8)))  float  v8f;

#define BB 4
#define SS 2048
#define DD 512
#define HH 8
#define HDD 64

// workspace byte offsets (ctxA aliases xA: xA dead after QKV GEMM)
#define OFF_XA    ((size_t)0)                    // 8 MB  bf16 x, A-frag packed (reused for ctx)
#define OFF_W     ((size_t)8388608)              // 2 MB  4x W, B-frag packed
#define OFF_QA    ((size_t)10485760)             // 8 MB  Q  A-frag per (b,h)
#define OFF_KB    ((size_t)18874368)             // 8 MB  K  B-frag per (b,h)
#define OFF_VB    ((size_t)27262976)             // 8 MB  V  B-frag per (b,h)
#define OFF_VSUM  ((size_t)35651584)             // 8 KB  f32 sum_k V
#define WMAT_ELS  262144                         // (512/32)*(512/16)*512 per matrix
#define BH_ELS    131072                         // per (b,h) fragment buffer elements

static __device__ __forceinline__ unsigned short bf16_bits(float f) {
  unsigned u = __float_as_uint(f);
  u += 0x7FFFu + ((u >> 16) & 1u);   // round-to-nearest-even
  return (unsigned short)(u >> 16);
}
static __device__ __forceinline__ __bf16 tobf(float f) {
  union { unsigned short s; __bf16 b; } c;
  c.s = bf16_bits(f);
  return c.b;
}
// K-index of element e in lane `lane` of a 16x32 bf16 A-fragment
static __device__ __forceinline__ int kOf(int lane, int e) {
  return ((e >> 3) << 4) + ((lane >> 4) << 3) + (e & 7);
}
static __device__ __forceinline__ v8f wmma_bf16(v16bf a, v16bf b, v8f c) {
  return __builtin_amdgcn_wmma_f32_16x16x32_bf16(false, a, false, b, (short)0, c, false, false);
}

// ---------------------------------------------------------------------------
// Pack x (8192x512 f32, row-major) into bf16 A-fragments (16x32 tiles)
// ---------------------------------------------------------------------------
__global__ void pack_x_kernel(const float* __restrict__ x,
                              unsigned short* __restrict__ xA) {
  int idx  = blockIdx.x * 256 + threadIdx.x;   // 4,194,304 total
  int f    = idx & 511;
  int tile = idx >> 9;
  int lane = f >> 4, e = f & 15;
  int tj = tile & 15, ti = tile >> 4;          // col-tile (d/32), row-tile (r/16)
  int r = (ti << 4) + (lane & 15);
  int d = (tj << 5) + kOf(lane, e);
  xA[idx] = bf16_bits(x[(size_t)r * DD + d]);
}

// ---------------------------------------------------------------------------
// Pack W (512x512, Y = X * W^T) into bf16 B-fragments (32(k=d) x 16(n=e))
// ---------------------------------------------------------------------------
__global__ void pack_w_kernel(const float* __restrict__ Wq, const float* __restrict__ Wk,
                              const float* __restrict__ Wv, const float* __restrict__ Wo,
                              unsigned short* __restrict__ Wp) {
  int idx = blockIdx.x * 256 + threadIdx.x;    // 262,144 per matrix
  int mi  = blockIdx.y;
  const float* W = (mi == 0) ? Wq : (mi == 1) ? Wk : (mi == 2) ? Wv : Wo;
  int f    = idx & 511;
  int tile = idx >> 9;
  int lane = f >> 4, e = f & 15;
  int tn = tile & 31, tk = tile >> 5;
  int d  = (tk << 5) + kOf(lane, e);           // k = d (table order)
  int ec = (tn << 4) + (lane & 15);            // n = output channel
  Wp[(size_t)mi * WMAT_ELS + idx] = bf16_bits(W[(size_t)ec * DD + d]);
}

// ---------------------------------------------------------------------------
// Fused Q/K/V projection GEMM, register-blocked 16x64 per wave (one head's
// columns). A fragment loaded once, fed to 4 WMMAs. Writes attention-native
// fragment layouts + V_sum.
// ---------------------------------------------------------------------------
__global__ void qkv_gemm_kernel(const unsigned short* __restrict__ xA,
                                const unsigned short* __restrict__ Wp,
                                const float* __restrict__ bq, const float* __restrict__ bk,
                                const float* __restrict__ bv,
                                unsigned short* __restrict__ Qa,
                                unsigned short* __restrict__ Kb,
                                unsigned short* __restrict__ Vb,
                                float* __restrict__ vsum) {
  int ti = blockIdx.x;          // row tile (0..511)
  int cg = blockIdx.y;          // column group of 64 == head (0..7)
  int mi = blockIdx.z;          // 0=Q 1=K 2=V
  int lane = threadIdx.x;
  const float* bias = (mi == 0) ? bq : (mi == 1) ? bk : bv;
  const unsigned short* wbase = Wp + (size_t)mi * WMAT_ELS;

  v8f acc[4];
#pragma unroll
  for (int c = 0; c < 4; ++c) acc[c] = (v8f){0.f,0.f,0.f,0.f,0.f,0.f,0.f,0.f};

#pragma unroll
  for (int kt = 0; kt < 16; ++kt) {
    v16bf a = *(const v16bf*)(xA + (size_t)(ti * 16 + kt) * 512 + lane * 16);
#pragma unroll
    for (int c = 0; c < 4; ++c) {
      v16bf b = *(const v16bf*)(wbase + (size_t)(kt * 32 + cg * 4 + c) * 512 + lane * 16);
      acc[c] = wmma_bf16(a, b, acc[c]);
    }
  }

  int n = lane & 15;
  int h = cg;
#pragma unroll
  for (int c = 0; c < 4; ++c) {
    int col = cg * 64 + c * 16 + n;
    int hd  = c * 16 + n;
    float bs = bias[col];
#pragma unroll
    for (int v = 0; v < 8; ++v) {
      int m  = v + ((lane >> 4) << 3);
      int r  = ti * 16 + m;
      int bi = r >> 11, s = r & 2047;
      float val = acc[c][v] + bs;
      unsigned short bits = bf16_bits(val);
      size_t bh = (size_t)(bi * HH + h);
      if (mi == 0) {                 // Q -> A-fragment layout per (b,h)
        int kk = hd & 31;
        int l2 = (s & 15) + (((kk >> 3) & 1) << 4);
        int e2 = ((kk >> 4) << 3) + (kk & 7);
        Qa[bh * BH_ELS + (size_t)(((s >> 4) << 1) + (hd >> 5)) * 512 + l2 * 16 + e2] = bits;
      } else if (mi == 1) {          // K -> B-fragment of K^T (64 x 2048)
        Kb[bh * BH_ELS + (size_t)(((hd >> 5) << 7) + (s >> 4)) * 512 + (((s & 15) + ((((hd & 31) >> 3) & 1) << 4)) << 4) + ((((hd & 31) >> 4) << 3) + (hd & 7))] = bits;
      } else {                       // V -> B-fragment (2048 x 64) + V_sum
        Vb[bh * BH_ELS + (size_t)(((s >> 5) << 2) + (hd >> 4)) * 512 + (((hd & 15) + ((((s & 31) >> 3) & 1) << 4)) << 4) + ((((s & 31) >> 4) << 3) + (s & 7))] = bits;
        atomicAdd(&vsum[(bi * HH + h) * 64 + hd], val);
      }
    }
  }
}

// ---------------------------------------------------------------------------
// Flash attention + docking blend, 32 queries (2 q-tiles) per wave:
// K and V fragments loaded once per 32-key step and reused by both q-tiles
// (16 WMMA : 16 global_load_b128).
// ctx = (1-beta)*softmax(QK^T/8, mask)*V + beta*ds[q]*V_sum
// ---------------------------------------------------------------------------
__global__ void attention_kernel(const unsigned short* __restrict__ Qa,
                                 const unsigned short* __restrict__ Kb,
                                 const unsigned short* __restrict__ Vb,
                                 const float* __restrict__ vsum,
                                 const float* __restrict__ dock,
                                 const int* __restrict__ mask,
                                 const float* __restrict__ betap,
                                 unsigned short* __restrict__ ctxA) {
  __shared__ __bf16 Pl[2 * 16 * 32];   // P tiles (layout shuffle), one per q-tile
  __shared__ __bf16 Cl[2 * 16 * 64];   // ctx staging
  __shared__ float  DSl[32];           // per-query docking score

  int qp = blockIdx.x;                 // q-tile pair (0..63)
  int h  = blockIdx.y, b = blockIdx.z;
  int qt0 = qp * 2;
  int lane = threadIdx.x;
  int n = lane & 15, g = lane >> 4;
  size_t bh = (size_t)(b * HH + h);
  const unsigned short* qb = Qa + bh * BH_ELS;
  const unsigned short* kb = Kb + bh * BH_ELS;
  const unsigned short* vb = Vb + bh * BH_ELS;
  const float beta = betap[0];

  v16bf aq[2][2];
#pragma unroll
  for (int qi = 0; qi < 2; ++qi)
#pragma unroll
    for (int d = 0; d < 2; ++d)
      aq[qi][d] = *(const v16bf*)(qb + (size_t)((qt0 + qi) * 2 + d) * 512 + lane * 16);

  v8f o[2][4];
  float mi_[2][8], li[2][8];
#pragma unroll
  for (int qi = 0; qi < 2; ++qi) {
#pragma unroll
    for (int c = 0; c < 4; ++c) o[qi][c] = (v8f){0.f,0.f,0.f,0.f,0.f,0.f,0.f,0.f};
#pragma unroll
    for (int v = 0; v < 8; ++v) { mi_[qi][v] = -1e30f; li[qi][v] = 0.f; }
  }

  for (int kt = 0; kt < 64; ++kt) {            // 32 keys per step
    int k2 = kt * 2;
    v16bf b00 = *(const v16bf*)(kb + (size_t)(k2)           * 512 + lane * 16);
    v16bf b10 = *(const v16bf*)(kb + (size_t)(128 + k2)     * 512 + lane * 16);
    v16bf b01 = *(const v16bf*)(kb + (size_t)(k2 + 1)       * 512 + lane * 16);
    v16bf b11 = *(const v16bf*)(kb + (size_t)(128 + k2 + 1) * 512 + lane * 16);

    v8f sc[2][2];
#pragma unroll
    for (int qi = 0; qi < 2; ++qi) {
      v8f z = (v8f){0.f,0.f,0.f,0.f,0.f,0.f,0.f,0.f};
      sc[qi][0] = wmma_bf16(aq[qi][0], b00, z);
      sc[qi][0] = wmma_bf16(aq[qi][1], b10, sc[qi][0]);
      sc[qi][1] = wmma_bf16(aq[qi][0], b01, z);
      sc[qi][1] = wmma_bf16(aq[qi][1], b11, sc[qi][1]);
    }

    bool mk0 = mask[b * SS + kt * 32 + n]      != 0;
    bool mk1 = mask[b * SS + kt * 32 + 16 + n] != 0;

    float corr[2][8];
#pragma unroll
    for (int qi = 0; qi < 2; ++qi) {
#pragma unroll
      for (int v = 0; v < 8; ++v) {
        float a0 = mk0 ? sc[qi][0][v] * 0.125f : -1e30f;
        float a1 = mk1 ? sc[qi][1][v] * 0.125f : -1e30f;
        float rm = fmaxf(a0, a1);
        rm = fmaxf(rm, __shfl_xor(rm, 1, 32));
        rm = fmaxf(rm, __shfl_xor(rm, 2, 32));
        rm = fmaxf(rm, __shfl_xor(rm, 4, 32));
        rm = fmaxf(rm, __shfl_xor(rm, 8, 32));   // row-wise over 16-lane half
        float mn = fmaxf(mi_[qi][v], rm);
        float c  = __expf(mi_[qi][v] - mn);
        float p0 = __expf(a0 - mn);
        float p1 = __expf(a1 - mn);
        float rs = p0 + p1;
        rs += __shfl_xor(rs, 1, 32);
        rs += __shfl_xor(rs, 2, 32);
        rs += __shfl_xor(rs, 4, 32);
        rs += __shfl_xor(rs, 8, 32);
        li[qi][v]  = li[qi][v] * c + rs;
        mi_[qi][v] = mn;
        corr[qi][v] = c;
        int row = v + (g << 3);
        Pl[qi * 512 + row * 32 + n]      = tobf(p0);
        Pl[qi * 512 + row * 32 + 16 + n] = tobf(p1);
      }
#pragma unroll
      for (int c = 0; c < 4; ++c)
#pragma unroll
        for (int v = 0; v < 8; ++v) o[qi][c][v] *= corr[qi][v];
    }
    __syncthreads();

    // D-fragment -> A-fragment layout shuffle via LDS (contiguous 16B runs)
    int prow = lane & 15;
    v16bf ap[2];
#pragma unroll
    for (int qi = 0; qi < 2; ++qi) {
      v8bf plo = *(const v8bf*)&Pl[qi * 512 + prow * 32 + (g << 3)];
      v8bf phi = *(const v8bf*)&Pl[qi * 512 + prow * 32 + 16 + (g << 3)];
#pragma unroll
      for (int e = 0; e < 8; ++e) { ap[qi][e] = plo[e]; ap[qi][e + 8] = phi[e]; }
    }

#pragma unroll
    for (int c = 0; c < 4; ++c) {
      v16bf vf = *(const v16bf*)(vb + (size_t)(kt * 4 + c) * 512 + lane * 16);
      o[0][c] = wmma_bf16(ap[0], vf, o[0][c]);
      o[1][c] = wmma_bf16(ap[1], vf, o[1][c]);
    }
    __syncthreads();

    if (kt < 63)
      __builtin_prefetch(kb + (size_t)(k2 + 2) * 512 + lane * 16, 0, 3);
  }

  // per-query docking scores (masked): 32 queries, one per lane
  {
    int q = b * SS + qt0 * 16 + lane;
    DSl[lane] = mask[q] ? dock[q] : 0.f;
  }
  __syncthreads();

  const float* vs = vsum + (b * HH + h) * 64;
  float ob = 1.f - beta;
#pragma unroll
  for (int qi = 0; qi < 2; ++qi) {
#pragma unroll
    for (int v = 0; v < 8; ++v) {
      int row = v + (g << 3);
      float inv = li[qi][v] > 0.f ? 1.f / li[qi][v] : 0.f;
      float bl  = beta * DSl[qi * 16 + row];
#pragma unroll
      for (int c = 0; c < 4; ++c) {
        float val = ob * o[qi][c][v] * inv + bl * vs[c * 16 + n];
        Cl[qi * 1024 + row * 64 + c * 16 + n] = tobf(val);
      }
    }
  }
  __syncthreads();

  // store ctx as packed A-fragments for the output GEMM
  int crow = lane & 15;
#pragma unroll
  for (int qi = 0; qi < 2; ++qi) {
    int rowTile = b * 128 + qt0 + qi;
#pragma unroll
    for (int kt = 0; kt < 2; ++kt) {
      v8bf lo = *(const v8bf*)&Cl[qi * 1024 + crow * 64 + kt * 32 + (g << 3)];
      v8bf hi = *(const v8bf*)&Cl[qi * 1024 + crow * 64 + kt * 32 + 16 + (g << 3)];
      v16bf cf;
#pragma unroll
      for (int e = 0; e < 8; ++e) { cf[e] = lo[e]; cf[e + 8] = hi[e]; }
      *(v16bf*)(ctxA + (size_t)(rowTile * 16 + h * 2 + kt) * 512 + lane * 16) = cf;
    }
  }
}

// ---------------------------------------------------------------------------
// Output projection: out = ctx * Wo^T + bo (f32 output), 16x64 per wave
// ---------------------------------------------------------------------------
__global__ void out_gemm_kernel(const unsigned short* __restrict__ ctxA,
                                const unsigned short* __restrict__ Wp,
                                const float* __restrict__ bo,
                                float* __restrict__ out) {
  int ti = blockIdx.x;          // row tile (0..511)
  int cg = blockIdx.y;          // column group of 64 (0..7)
  int lane = threadIdx.x;
  const unsigned short* wbase = Wp + (size_t)3 * WMAT_ELS;

  v8f acc[4];
#pragma unroll
  for (int c = 0; c < 4; ++c) acc[c] = (v8f){0.f,0.f,0.f,0.f,0.f,0.f,0.f,0.f};

#pragma unroll
  for (int kt = 0; kt < 16; ++kt) {
    v16bf a = *(const v16bf*)(ctxA + (size_t)(ti * 16 + kt) * 512 + lane * 16);
#pragma unroll
    for (int c = 0; c < 4; ++c) {
      v16bf b = *(const v16bf*)(wbase + (size_t)(kt * 32 + cg * 4 + c) * 512 + lane * 16);
      acc[c] = wmma_bf16(a, b, acc[c]);
    }
  }
  int n = lane & 15, g = lane >> 4;
#pragma unroll
  for (int c = 0; c < 4; ++c) {
    int col = cg * 64 + c * 16 + n;
    float bs = bo[col];
#pragma unroll
    for (int v = 0; v < 8; ++v) {
      int r = ti * 16 + v + (g << 3);
      out[(size_t)r * DD + col] = acc[c][v] + bs;
    }
  }
}

// ---------------------------------------------------------------------------
extern "C" void kernel_launch(void* const* d_in, const int* in_sizes, int n_in,
                              void* d_out, int out_size, void* d_ws, size_t ws_size,
                              hipStream_t stream) {
  const float* x    = (const float*)d_in[0];
  const float* dock = (const float*)d_in[1];
  const int*   mask = (const int*)d_in[2];
  const float* Wq   = (const float*)d_in[3];
  const float* bq   = (const float*)d_in[4];
  const float* Wk   = (const float*)d_in[5];
  const float* bk   = (const float*)d_in[6];
  const float* Wv   = (const float*)d_in[7];
  const float* bv   = (const float*)d_in[8];
  const float* Wo   = (const float*)d_in[9];
  const float* bo   = (const float*)d_in[10];
  // d_in[11] = alpha (unused in ATTENTION path)
  const float* beta = (const float*)d_in[12];
  float* out = (float*)d_out;

  char* ws = (char*)d_ws;
  unsigned short* xA   = (unsigned short*)(ws + OFF_XA);
  unsigned short* Wp   = (unsigned short*)(ws + OFF_W);
  unsigned short* Qa   = (unsigned short*)(ws + OFF_QA);
  unsigned short* Kb   = (unsigned short*)(ws + OFF_KB);
  unsigned short* Vb   = (unsigned short*)(ws + OFF_VB);
  float*          vsum = (float*)(ws + OFF_VSUM);
  unsigned short* ctxA = (unsigned short*)(ws + OFF_XA);  // reuse x-pack region

  hipMemsetAsync(vsum, 0, BB * HH * HDD * sizeof(float), stream);

  pack_x_kernel<<<16384, 256, 0, stream>>>(x, xA);
  pack_w_kernel<<<dim3(1024, 4), 256, 0, stream>>>(Wq, Wk, Wv, Wo, Wp);
  qkv_gemm_kernel<<<dim3(512, 8, 3), 32, 0, stream>>>(xA, Wp, bq, bk, bv, Qa, Kb, Vb, vsum);
  attention_kernel<<<dim3(64, 8, 4), 32, 0, stream>>>(Qa, Kb, Vb, vsum, dock, mask, beta, ctxA);
  out_gemm_kernel<<<dim3(512, 8), 32, 0, stream>>>(ctxA, Wp, bo, out);
}
